// EGNNLayer_64759516889097
// MI455X (gfx1250) — matrix operand order repaired and checked by reading, not compiled, not run on hardware
//
#include <hip/hip_runtime.h>
#include <math.h>

// ---------------------------------------------------------------------------
// MI455X (gfx1250) implementation of the EGNN layer.
// Both tensor-product stages are refactored into GEMMs over outer-product
// feature vectors and executed with v_wmma_f32_16x16x32_bf16 (wave32,
// M=16 edges/nodes per wave, fp32 accumulate). Weights are pre-converted
// once per launch into fragment-major bf16 so B-operand loads are a single
// contiguous 32B load per lane. A-operands are built directly in registers
// in the ISA-documented 16-bit A layout (lane<16: K 0-7,16-23; lane>=16:
// K 8-15,24-31), so no LDS staging is needed on the A path.
// ---------------------------------------------------------------------------

typedef __bf16 bf16;
typedef __attribute__((ext_vector_type(16))) __bf16 v16bf;
typedef __attribute__((ext_vector_type(8)))  __bf16 v8bf;
typedef __attribute__((ext_vector_type(8)))  float  v8f;

#define N_NODES 20000
#define N_EDGES 400000
#define NS 64
#define NV 32
#define ES 8
#define EV 8
#define HS 96
#define HV 32
#define GS 64
#define NF_DIM 160   // NS + 3*NV
#define EA_DIM 32    // ES + 3*EV

#define KM   768     // message-stage combined K (hs and hv)
#define KU_S 7168    // update-stage hs K = 64*96 + 32*32
#define KU_V 5120    // update-stage hv K = 64*32 + 32*96

__device__ __forceinline__ bf16 f2bf(float f) { return (bf16)f; }

__device__ __forceinline__ v8f v8f_zero() {
  v8f z = {0.f, 0.f, 0.f, 0.f, 0.f, 0.f, 0.f, 0.f};
  return z;
}

__device__ __forceinline__ v8f wmma_bf16(v16bf a, v16bf b, v8f c) {
  // D = A(16x32) * B(32x16) + C(16x16), fp32 accumulate
  return __builtin_amdgcn_wmma_f32_16x16x32_bf16(
      false, a, false, b, (short)0, c, false, false);
}

// A-fragment K index (within a 32-wide chunk) held at (lane, element j):
// lane<16 holds K 0-7 (j=0..7) and 16-23 (j=8..15); lane>=16 holds 8-15, 24-31.
__device__ __forceinline__ int a_kmap(int lane, int j) {
  return ((j & 8) << 1) + ((lane >> 4) << 3) + (j & 7);
}

// ---------------------------------------------------------------------------
// Weight prep: combined [K][Nout] fp32 matrix (two stacked source tensors,
// per-part scale) -> bf16 in WMMA-B fragment-major order:
//   tile (kt,nt) : 512 elems; lane = ((k%32 / 8)%2)*16 + (n%16); j = k%8 + (k%32 / 16)*8.
// ---------------------------------------------------------------------------
__global__ void prep_combined(const float* __restrict__ W1,
                              const float* __restrict__ W2,
                              bf16* __restrict__ dst,
                              int K1, int K, int Nout, float s1, float s2) {
  int tid = blockIdx.x * blockDim.x + threadIdx.x;
  if (tid >= K * Nout) return;
  int k = tid / Nout;
  int w = tid - k * Nout;
  float v = (k < K1) ? W1[(size_t)k * Nout + w] * s1
                     : W2[(size_t)(k - K1) * Nout + w] * s2;
  int kt = k >> 5, kk = k & 31, nt = w >> 4, n = w & 15;
  int lanei = (((kk >> 3) & 1) << 4) + n;
  int j = (kk & 7) + ((kk >> 4) << 3);
  dst[((size_t)(kt * (Nout >> 4) + nt)) * 512 + lanei * 16 + j] = f2bf(v);
}

__global__ void zero_f32(float* __restrict__ p, long long n) {
  long long i = (long long)blockIdx.x * blockDim.x + threadIdx.x;
  long long stride = (long long)gridDim.x * blockDim.x;
  for (; i < n; i += stride) p[i] = 0.0f;
}

// ---------------------------------------------------------------------------
// Message kernel: one wave32 per 16-edge tile.
//   hs[16,96]  = Z[16,768]   @ Whs[768,96]    (WMMA, 24 K-chunks x 6 N-tiles)
//   hv_c[16,32]= F_c[16,768] @ Whv[768,32]    (x3 components)
//   gate, then ms = gs@Wlms (K=64,N=96), mv_c = gv_c@Wlmv (K=32,N=32)
//   atomic scatter-add into ags[N,96], agv[N,32,3]
// ---------------------------------------------------------------------------
__global__ __launch_bounds__(32) void msg_kernel(
    const float* __restrict__ nf, const float* __restrict__ ea,
    const int* __restrict__ eidx,
    const bf16* __restrict__ Whs, const bf16* __restrict__ Whv,
    const bf16* __restrict__ Wlms, const bf16* __restrict__ Wlmv,
    float* __restrict__ ags, float* __restrict__ agv) {
  const int lane = threadIdx.x;
  const int e0 = blockIdx.x * 16;

  __shared__ float s_xs[16][NS];
  __shared__ float s_xv[16][3 * NV];
  __shared__ float s_es[16][ES];
  __shared__ float s_ev[16][3 * EV];
  __shared__ int s_row[16];
  __shared__ int s_col[16];
  __shared__ float s_gate[16][HV];
  __shared__ __align__(32) bf16 s_gs[16 * GS];
  __shared__ __align__(32) bf16 s_gv[3][16 * HV];

  if (lane < 16) {
    s_row[lane] = eidx[e0 + lane];            // edge_index[0] = dst rows
    s_col[lane] = eidx[N_EDGES + e0 + lane];  // edge_index[1] = src cols
  }
  __syncthreads();

  // Gather source-node features + edge attrs into LDS (fp32).
  for (int e = 0; e < 16; ++e) {
    const float* src = nf + (size_t)s_col[e] * NF_DIM;
    for (int i = lane; i < NF_DIM; i += 32) {
      float v = src[i];
      if (i < NS) s_xs[e][i] = v; else s_xv[e][i - NS] = v;
    }
    float v = ea[(size_t)(e0 + e) * EA_DIM + lane];
    if (lane < ES) s_es[e][lane] = v; else s_ev[e][lane - ES] = v;
  }
  __syncthreads();

  const int er = lane & 15;     // A-matrix row (edge in tile)
  const int half = lane >> 4;
  const int cn = lane & 15;     // C-matrix column

  v8f acc_hs[6], acc_hv[6];
  for (int i = 0; i < 6; ++i) { acc_hs[i] = v8f_zero(); acc_hv[i] = v8f_zero(); }

  for (int kc = 0; kc < KM / 32; ++kc) {
    v16bf a_hs, a_v0, a_v1, a_v2;
#pragma unroll
    for (int j = 0; j < 16; ++j) {
      int kg = kc * 32 + a_kmap(lane, j);
      float fhs, f0, f1, f2;
      if (kg < NS * ES) {                 // scalar x scalar / scalar x vector
        int u = kg >> 3, vv = kg & 7;
        float x = s_xs[er][u];
        fhs = x * s_es[er][vv];
        f0 = x * s_ev[er][vv * 3 + 0];
        f1 = x * s_ev[er][vv * 3 + 1];
        f2 = x * s_ev[er][vv * 3 + 2];
      } else {                            // vector.vector / vector x scalar
        int idx = kg - NS * ES;
        int u = idx >> 3, vv = idx & 7;
        float x0 = s_xv[er][u * 3 + 0], x1 = s_xv[er][u * 3 + 1], x2 = s_xv[er][u * 3 + 2];
        fhs = x0 * s_ev[er][vv * 3 + 0] + x1 * s_ev[er][vv * 3 + 1] + x2 * s_ev[er][vv * 3 + 2];
        float sv = s_es[er][vv];
        f0 = x0 * sv; f1 = x1 * sv; f2 = x2 * sv;
      }
      a_hs[j] = f2bf(fhs); a_v0[j] = f2bf(f0); a_v1[j] = f2bf(f1); a_v2[j] = f2bf(f2);
    }
    const bf16* bs = Whs + (size_t)kc * 6 * 512 + lane * 16;
#pragma unroll
    for (int nt = 0; nt < 6; ++nt) {
      v16bf b = *(const v16bf*)(bs + nt * 512);
      acc_hs[nt] = wmma_bf16(a_hs, b, acc_hs[nt]);
    }
    const bf16* bv = Whv + (size_t)kc * 2 * 512 + lane * 16;
#pragma unroll
    for (int nt = 0; nt < 2; ++nt) {
      v16bf b = *(const v16bf*)(bv + nt * 512);
      acc_hv[0 + nt] = wmma_bf16(a_v0, b, acc_hv[0 + nt]);
      acc_hv[2 + nt] = wmma_bf16(a_v1, b, acc_hv[2 + nt]);
      acc_hv[4 + nt] = wmma_bf16(a_v2, b, acc_hv[4 + nt]);
    }
  }

  // Gate: silu on hs[:, :64], sigmoid gates from hs[:, 64:96].
#pragma unroll
  for (int nt = 0; nt < 6; ++nt) {
#pragma unroll
    for (int r = 0; r < 8; ++r) {
      int e = r + half * 8;
      int w = nt * 16 + cn;
      float h = acc_hs[nt][r];
      float sg = 1.0f / (1.0f + __expf(-h));
      if (w < GS) s_gs[e * GS + w] = f2bf(h * sg);
      else s_gate[e][w - GS] = sg;
    }
  }
  __syncthreads();
#pragma unroll
  for (int c = 0; c < 3; ++c)
#pragma unroll
    for (int nt = 0; nt < 2; ++nt)
#pragma unroll
      for (int r = 0; r < 8; ++r) {
        int e = r + half * 8;
        int w = nt * 16 + cn;
        s_gv[c][e * HV + w] = f2bf(acc_hv[c * 2 + nt][r] * s_gate[e][w]);
      }
  __syncthreads();

  // Linear: ms = gs @ Wlms (K=64,N=96); mv_c = gv_c @ Wlmv (K=32,N=32).
  v8f ms[6];
  for (int i = 0; i < 6; ++i) ms[i] = v8f_zero();
#pragma unroll
  for (int kc = 0; kc < 2; ++kc) {
    int kb = kc * 32 + half * 8;
    v8bf lo = *(const v8bf*)&s_gs[er * GS + kb];
    v8bf hi = *(const v8bf*)&s_gs[er * GS + kb + 16];
    v16bf a;
#pragma unroll
    for (int j = 0; j < 8; ++j) { a[j] = lo[j]; a[j + 8] = hi[j]; }
    const bf16* bs = Wlms + (size_t)kc * 6 * 512 + lane * 16;
#pragma unroll
    for (int nt = 0; nt < 6; ++nt) {
      v16bf b = *(const v16bf*)(bs + nt * 512);
      ms[nt] = wmma_bf16(a, b, ms[nt]);
    }
  }
  v8f mv[6];
  for (int i = 0; i < 6; ++i) mv[i] = v8f_zero();
#pragma unroll
  for (int c = 0; c < 3; ++c) {
    int kb = half * 8;
    v8bf lo = *(const v8bf*)&s_gv[c][er * HV + kb];
    v8bf hi = *(const v8bf*)&s_gv[c][er * HV + kb + 16];
    v16bf a;
#pragma unroll
    for (int j = 0; j < 8; ++j) { a[j] = lo[j]; a[j + 8] = hi[j]; }
    const bf16* bs = Wlmv + lane * 16;
#pragma unroll
    for (int nt = 0; nt < 2; ++nt) {
      v16bf b = *(const v16bf*)(bs + nt * 512);
      mv[c * 2 + nt] = wmma_bf16(a, b, mv[c * 2 + nt]);
    }
  }

  // Scatter-add (segment_sum) into per-node aggregation buffers.
#pragma unroll
  for (int nt = 0; nt < 6; ++nt)
#pragma unroll
    for (int r = 0; r < 8; ++r) {
      int e = r + half * 8;
      int w = nt * 16 + cn;
      atomicAdd(&ags[(size_t)s_row[e] * HS + w], ms[nt][r]);
    }
#pragma unroll
  for (int c = 0; c < 3; ++c)
#pragma unroll
    for (int nt = 0; nt < 2; ++nt)
#pragma unroll
      for (int r = 0; r < 8; ++r) {
        int e = r + half * 8;
        int w = nt * 16 + cn;
        atomicAdd(&agv[(size_t)s_row[e] * (HV * 3) + w * 3 + c], mv[c * 2 + nt][r]);
      }
}

// ---------------------------------------------------------------------------
// Update kernel: one wave32 per 16-node tile. Same GEMM structure with
// K=7168 (hs) and K=5120 (hv per component), then gate, final linears,
// residual add, store to d_out.
// ---------------------------------------------------------------------------
__global__ __launch_bounds__(32) void upd_kernel(
    const float* __restrict__ nf,
    const float* __restrict__ ags, const float* __restrict__ agv,
    const bf16* __restrict__ Whs, const bf16* __restrict__ Whv,
    const bf16* __restrict__ Wlus, const bf16* __restrict__ Wluv,
    float* __restrict__ out) {
  const int lane = threadIdx.x;
  const int n0 = blockIdx.x * 16;

  __shared__ float s_xs[16][NS];
  __shared__ float s_xv[16][3 * NV];
  __shared__ float s_ys[16][HS];
  __shared__ float s_yv[16][3 * HV];
  __shared__ float s_gate[16][HV];
  __shared__ __align__(32) bf16 s_gs[16 * GS];
  __shared__ __align__(32) bf16 s_gv[3][16 * HV];

  for (int e = 0; e < 16; ++e) {
    size_t nid = (size_t)(n0 + e);
    const float* src = nf + nid * NF_DIM;
    for (int i = lane; i < NF_DIM; i += 32) {
      float v = src[i];
      if (i < NS) s_xs[e][i] = v; else s_xv[e][i - NS] = v;
    }
    for (int i = lane; i < HS; i += 32) s_ys[e][i] = ags[nid * HS + i];
    for (int i = lane; i < 3 * HV; i += 32) s_yv[e][i] = agv[nid * (3 * HV) + i];
  }
  __syncthreads();

  const int er = lane & 15;
  const int half = lane >> 4;
  const int cn = lane & 15;

  v8f acc_hs[6];
  for (int i = 0; i < 6; ++i) acc_hs[i] = v8f_zero();
  for (int kc = 0; kc < KU_S / 32; ++kc) {  // 224 chunks
    v16bf a;
#pragma unroll
    for (int j = 0; j < 16; ++j) {
      int kg = kc * 32 + a_kmap(lane, j);
      float f;
      if (kg < NS * HS) {                       // xs (64) x ags (96)
        int u = kg / HS, vv = kg - (kg / HS) * HS;
        f = s_xs[er][u] * s_ys[er][vv];
      } else {                                  // xv (32) . agv (32) 3-dot
        int idx = kg - NS * HS;
        int u = idx >> 5, vv = idx & 31;
        f = s_xv[er][u * 3 + 0] * s_yv[er][vv * 3 + 0] +
            s_xv[er][u * 3 + 1] * s_yv[er][vv * 3 + 1] +
            s_xv[er][u * 3 + 2] * s_yv[er][vv * 3 + 2];
      }
      a[j] = f2bf(f);
    }
    const bf16* bs = Whs + (size_t)kc * 6 * 512 + lane * 16;
    __builtin_prefetch((const void*)(bs + 6 * 512), 0, 1);  // global_prefetch next K-chunk
#pragma unroll
    for (int nt = 0; nt < 6; ++nt) {
      v16bf b = *(const v16bf*)(bs + nt * 512);
      acc_hs[nt] = wmma_bf16(a, b, acc_hs[nt]);
    }
  }

  v8f acc_hv[6];
  for (int i = 0; i < 6; ++i) acc_hv[i] = v8f_zero();
  for (int kc = 0; kc < KU_V / 32; ++kc) {  // 160 chunks
    v16bf a0, a1, a2;
#pragma unroll
    for (int j = 0; j < 16; ++j) {
      int kg = kc * 32 + a_kmap(lane, j);
      float f0, f1, f2;
      if (kg < NS * HV) {                       // xs (64) x agv (32)
        int u = kg >> 5, vv = kg & 31;
        float x = s_xs[er][u];
        f0 = x * s_yv[er][vv * 3 + 0];
        f1 = x * s_yv[er][vv * 3 + 1];
        f2 = x * s_yv[er][vv * 3 + 2];
      } else {                                  // xv (32) x ags (96)
        int idx = kg - NS * HV;
        int u = idx / HS, vv = idx - (idx / HS) * HS;
        float sv = s_ys[er][vv];
        f0 = s_xv[er][u * 3 + 0] * sv;
        f1 = s_xv[er][u * 3 + 1] * sv;
        f2 = s_xv[er][u * 3 + 2] * sv;
      }
      a0[j] = f2bf(f0); a1[j] = f2bf(f1); a2[j] = f2bf(f2);
    }
    const bf16* bv = Whv + (size_t)kc * 2 * 512 + lane * 16;
#pragma unroll
    for (int nt = 0; nt < 2; ++nt) {
      v16bf b = *(const v16bf*)(bv + nt * 512);
      acc_hv[0 + nt] = wmma_bf16(a0, b, acc_hv[0 + nt]);
      acc_hv[2 + nt] = wmma_bf16(a1, b, acc_hv[2 + nt]);
      acc_hv[4 + nt] = wmma_bf16(a2, b, acc_hv[4 + nt]);
    }
  }

  // Gate.
#pragma unroll
  for (int nt = 0; nt < 6; ++nt)
#pragma unroll
    for (int r = 0; r < 8; ++r) {
      int e = r + half * 8;
      int w = nt * 16 + cn;
      float h = acc_hs[nt][r];
      float sg = 1.0f / (1.0f + __expf(-h));
      if (w < GS) s_gs[e * GS + w] = f2bf(h * sg);
      else s_gate[e][w - GS] = sg;
    }
  __syncthreads();
#pragma unroll
  for (int c = 0; c < 3; ++c)
#pragma unroll
    for (int nt = 0; nt < 2; ++nt)
#pragma unroll
      for (int r = 0; r < 8; ++r) {
        int e = r + half * 8;
        int w = nt * 16 + cn;
        s_gv[c][e * HV + w] = f2bf(acc_hv[c * 2 + nt][r] * s_gate[e][w]);
      }
  __syncthreads();

  // Final linears: us = gs @ Wlus (K=64,N=64); uv_c = gv_c @ Wluv (K=32,N=32).
  v8f us[4];
  for (int i = 0; i < 4; ++i) us[i] = v8f_zero();
#pragma unroll
  for (int kc = 0; kc < 2; ++kc) {
    int kb = kc * 32 + half * 8;
    v8bf lo = *(const v8bf*)&s_gs[er * GS + kb];
    v8bf hi = *(const v8bf*)&s_gs[er * GS + kb + 16];
    v16bf a;
#pragma unroll
    for (int j = 0; j < 8; ++j) { a[j] = lo[j]; a[j + 8] = hi[j]; }
    const bf16* bs = Wlus + (size_t)kc * 4 * 512 + lane * 16;
#pragma unroll
    for (int nt = 0; nt < 4; ++nt) {
      v16bf b = *(const v16bf*)(bs + nt * 512);
      us[nt] = wmma_bf16(a, b, us[nt]);
    }
  }
  v8f uv[6];
  for (int i = 0; i < 6; ++i) uv[i] = v8f_zero();
#pragma unroll
  for (int c = 0; c < 3; ++c) {
    int kb = half * 8;
    v8bf lo = *(const v8bf*)&s_gv[c][er * HV + kb];
    v8bf hi = *(const v8bf*)&s_gv[c][er * HV + kb + 16];
    v16bf a;
#pragma unroll
    for (int j = 0; j < 8; ++j) { a[j] = lo[j]; a[j + 8] = hi[j]; }
    const bf16* bs = Wluv + lane * 16;
#pragma unroll
    for (int nt = 0; nt < 2; ++nt) {
      v16bf b = *(const v16bf*)(bs + nt * 512);
      uv[c * 2 + nt] = wmma_bf16(a, b, uv[c * 2 + nt]);
    }
  }

  // Residual add + store: out = node_features + [us (64) | uv (32x3)].
#pragma unroll
  for (int nt = 0; nt < 4; ++nt)
#pragma unroll
    for (int r = 0; r < 8; ++r) {
      int e = r + half * 8;
      int w = nt * 16 + cn;
      size_t o = (size_t)(n0 + e) * NF_DIM + w;
      out[o] = nf[o] + us[nt][r];
    }
#pragma unroll
  for (int c = 0; c < 3; ++c)
#pragma unroll
    for (int nt = 0; nt < 2; ++nt)
#pragma unroll
      for (int r = 0; r < 8; ++r) {
        int e = r + half * 8;
        int w = nt * 16 + cn;
        size_t o = (size_t)(n0 + e) * NF_DIM + NS + w * 3 + c;
        out[o] = nf[o] + uv[c * 2 + nt][r];
      }
}

// ---------------------------------------------------------------------------
extern "C" void kernel_launch(void* const* d_in, const int* in_sizes, int n_in,
                              void* d_out, int out_size, void* d_ws, size_t ws_size,
                              hipStream_t stream) {
  (void)in_sizes; (void)n_in; (void)out_size; (void)ws_size;

  const float* nf   = (const float*)d_in[0];
  const float* ea   = (const float*)d_in[1];
  const int*   eidx = (const int*)d_in[3];   // d_in[2] (node_attr_scalar_raw) unused by reference
  const float* Wmss = (const float*)d_in[4];
  const float* Wmvv = (const float*)d_in[5];
  const float* Wmsv = (const float*)d_in[6];
  const float* Wmvs = (const float*)d_in[7];
  const float* Wlms = (const float*)d_in[8];
  const float* Wlmv = (const float*)d_in[9];
  const float* Wuss = (const float*)d_in[10];
  const float* Wuvv = (const float*)d_in[11];
  const float* Wusv = (const float*)d_in[12];
  const float* Wuvs = (const float*)d_in[13];
  const float* Wlus = (const float*)d_in[14];
  const float* Wluv = (const float*)d_in[15];
  float* out = (float*)d_out;
  char* ws = (char*)d_ws;

  size_t off = 0;
  auto take = [&](size_t bytes) {
    size_t o = off;
    off = (off + bytes + 255) & ~(size_t)255;
    return o;
  };
  bf16* bWhsM = (bf16*)(ws + take((size_t)KM * HS * 2));     // 768x96
  bf16* bWhvM = (bf16*)(ws + take((size_t)KM * HV * 2));     // 768x32
  bf16* bWlms = (bf16*)(ws + take((size_t)GS * HS * 2));     // 64x96
  bf16* bWlmv = (bf16*)(ws + take((size_t)HV * HV * 2));     // 32x32
  bf16* bWhsU = (bf16*)(ws + take((size_t)KU_S * HS * 2));   // 7168x96
  bf16* bWhvU = (bf16*)(ws + take((size_t)KU_V * HV * 2));   // 5120x32
  bf16* bWlus = (bf16*)(ws + take((size_t)GS * NS * 2));     // 64x64
  bf16* bWluv = (bf16*)(ws + take((size_t)HV * NV * 2));     // 32x32
  float* ags = (float*)(ws + take((size_t)N_NODES * HS * 4));
  float* agv = (float*)(ws + take((size_t)N_NODES * HV * 3 * 4));

  const float INV_SQRT3 = 0.5773502691896258f;
  const float nsm = 1.0f / sqrtf((float)(NS * ES + NV * EV));   // 1/sqrt(768)
  const float nvm = 1.0f / sqrtf((float)(NS * EV + NV * ES));   // 1/sqrt(768)
  const float nsu = 1.0f / sqrtf((float)(NS * HS + NV * HV));   // 1/sqrt(7168)
  const float nvu = 1.0f / sqrtf((float)(NS * HV + NV * HS));   // 1/sqrt(5120)
  const float l64 = 1.0f / sqrtf(64.0f);
  const float l32 = 1.0f / sqrtf(32.0f);

  auto prep = [&](const float* W1, const float* W2, bf16* dst,
                  int K1, int K, int Nout, float s1, float s2) {
    int total = K * Nout;
    prep_combined<<<(total + 255) / 256, 256, 0, stream>>>(W1, W2, dst, K1, K, Nout, s1, s2);
  };
  prep(Wmss, Wmvv, bWhsM, NS * ES, KM,   HS, nsm, nsm * INV_SQRT3);
  prep(Wmsv, Wmvs, bWhvM, NS * EV, KM,   HV, nvm, nvm);
  prep(Wlms, Wlms, bWlms, GS,      GS,   HS, l64, l64);
  prep(Wlmv, Wlmv, bWlmv, HV,      HV,   HV, l32, l32);
  prep(Wuss, Wuvv, bWhsU, NS * HS, KU_S, HS, nsu, nsu * INV_SQRT3);
  prep(Wusv, Wuvs, bWhvU, NS * HV, KU_V, HV, nvu, nvu);
  prep(Wlus, Wlus, bWlus, GS,      GS,   NS, l64, l64);
  prep(Wluv, Wluv, bWluv, HV,      HV,   NV, l32, l32);

  zero_f32<<<2048, 256, 0, stream>>>(ags, (long long)N_NODES * HS);
  zero_f32<<<2048, 256, 0, stream>>>(agv, (long long)N_NODES * HV * 3);

  msg_kernel<<<N_EDGES / 16, 32, 0, stream>>>(nf, ea, eidx, bWhsM, bWhvM, bWlms, bWlmv, ags, agv);
  upd_kernel<<<N_NODES / 16, 32, 0, stream>>>(nf, ags, agv, bWhsU, bWhvU, bWlus, bWluv, out);
}
